// PillarLayer_57071525429695
// MI455X (gfx1250) — compile-verified, matrix operand order checked by reading, not run on hardware
//
#include <hip/hip_runtime.h>
#include <hip/hip_bf16.h>
#include <stdint.h>

#define B_       4
#define N_       120000
#define C_       4
#define MAX_PTS_ 32
#define MAX_VOX_ 40000
#define GX_      432
#define GY_      496
#define GZ_      1
#define NVOX_    (GX_ * GY_ * GZ_)   // 214272

typedef __attribute__((ext_vector_type(16))) _Float16 v16h;
typedef __attribute__((ext_vector_type(8)))  float    v8f;

// ---------------- output init: pillars=0, coords=-1, npts=0 ----------------
__global__ void init_out_kernel(float* __restrict__ out) {
    const long long PILL = (long long)B_ * MAX_VOX_ * MAX_PTS_ * C_;   // 20,480,000
    const long long COOR = PILL + (long long)B_ * MAX_VOX_ * 4;        // +640,000
    const long long TOT  = COOR + (long long)B_ * MAX_VOX_;            // +160,000
    for (long long i = (long long)blockIdx.x * blockDim.x + threadIdx.x; i < TOT;
         i += (long long)gridDim.x * blockDim.x) {
        if (i < PILL)       out[i] = 0.0f;
        else if (i < COOR)  ((int*)out)[i] = -1;
        else                ((int*)out)[i] = 0;
    }
}

// ---------------- workspace init ----------------
__global__ void init_ws_kernel(int* __restrict__ firstp, int* __restrict__ cnt,
                               int* __restrict__ slot,   int* __restrict__ tmpm) {
    const int total = B_ * NVOX_;
    for (int i = blockIdx.x * blockDim.x + threadIdx.x; i < total;
         i += gridDim.x * blockDim.x) {
        firstp[i] = 0x7FFFFFFF;
        tmpm[i]   = 0x7FFFFFFF;
        cnt[i]    = 0;
        slot[i]   = -1;
    }
}

// ---------------- per-point flat voxel id + first-point / count atomics ----
// Stages the point tile into LDS with CDNA5 async-copy (ASYNCcnt path).
__global__ __launch_bounds__(256) void flat_kernel(const float* __restrict__ pts,
                                                   int* __restrict__ flat,
                                                   int* __restrict__ rank,
                                                   int* __restrict__ firstp,
                                                   int* __restrict__ cnt) {
    __shared__ float spts[256 * 4] __attribute__((aligned(16)));
    const int tid = threadIdx.x;
    const long long idx = (long long)blockIdx.x * 256 + tid;   // B_*N_ = 480000 = 1875*256 exact

    const float* src = pts + idx * 4;
    __builtin_prefetch(src + 4096, 0, 1);                      // global_prefetch_b8

    uint32_t ldsoff = (uint32_t)(uintptr_t)(&spts[tid * 4]);   // low 32 bits of flat shared addr = LDS offset
    uint64_t gaddr  = (uint64_t)(uintptr_t)src;
    asm volatile("global_load_async_to_lds_b128 %0, %1, off"
                 :: "v"(ldsoff), "v"(gaddr) : "memory");
    asm volatile("s_wait_asynccnt 0" ::: "memory");
    __syncthreads();

    const float x = spts[tid * 4 + 0];
    const float y = spts[tid * 4 + 1];
    const float z = spts[tid * 4 + 2];

    const int b = (int)(idx / N_);
    const int p = (int)(idx - (long long)b * N_);

    const bool inb = (x >= 0.0f) && (x < 69.12f) &&
                     (y >= -39.68f) && (y < 39.68f) &&
                     (z >= -3.0f) && (z < 1.0f);
    int f = -1;
    if (inb) {
        int ix = (int)floorf((x - 0.0f)   / 0.16f);
        int iy = (int)floorf((y + 39.68f) / 0.16f);
        int iz = (int)floorf((z + 3.0f)   / 4.0f);
        ix = ix < 0 ? 0 : (ix > GX_ - 1 ? GX_ - 1 : ix);
        iy = iy < 0 ? 0 : (iy > GY_ - 1 ? GY_ - 1 : iy);
        iz = iz < 0 ? 0 : (iz > GZ_ - 1 ? GZ_ - 1 : iz);
        f = (ix * GY_ + iy) * GZ_ + iz;
        atomicMin(&firstp[b * NVOX_ + f], p);
        atomicAdd(&cnt[b * NVOX_ + f], 1);
    }
    flat[idx] = f;
    rank[idx] = -1;
}

// ---------------- per-batch slot assignment (prefix scan of isFirst) ------
// Chunk totals for the scan carry are computed with v_wmma_f32_16x16x32_f16:
// with A = all-ones, sum(D) = 16 * sum(B) independent of fragment layout.
__global__ __launch_bounds__(1024) void slot_kernel(const int* __restrict__ flat,
                                                    const int* __restrict__ firstp,
                                                    int* __restrict__ slot) {
    __shared__ _Float16 sflags[1024] __attribute__((aligned(32)));
    __shared__ float swave[32];
    __shared__ float schunk;

    const int b    = blockIdx.x;
    const int tid  = threadIdx.x;
    const int lane = tid & 31;
    const int wave = tid >> 5;

    v16h ones;
#pragma unroll
    for (int i = 0; i < 16; ++i) ones[i] = (_Float16)1.0f;

    int carry = 0;
    const int NCH = (N_ + 1023) / 1024;
    for (int ch = 0; ch < NCH; ++ch) {
        const int p  = ch * 1024 + tid;
        const int fl = (p < N_) ? flat[(long long)b * N_ + p] : -1;
        const int flag = (fl >= 0 && firstp[b * NVOX_ + fl] == p) ? 1 : 0;

        // wave32 inclusive scan
        int incl = flag;
#pragma unroll
        for (int off = 1; off < 32; off <<= 1) {
            int nv = __shfl_up(incl, off, 32);
            if (lane >= off) incl += nv;
        }

        sflags[tid] = (_Float16)(float)flag;
        if (lane == 31) swave[wave] = (float)incl;
        __syncthreads();

        if (wave == 0) {                       // exclusive scan of 32 wave totals
            float wv = swave[lane];
            float wi = wv;
#pragma unroll
            for (int off = 1; off < 32; off <<= 1) {
                float nv = __shfl_up(wi, off, 32);
                if (lane >= off) wi += nv;
            }
            swave[lane] = wi - wv;
        } else if (wave == 1) {                // chunk total via matrix engine
            const v16h* bp = (const v16h*)sflags;
            v8f c = {};
            c = __builtin_amdgcn_wmma_f32_16x16x32_f16(false, ones, false, bp[lane],
                                                       (short)0, c, false, false);
            c = __builtin_amdgcn_wmma_f32_16x16x32_f16(false, ones, false, bp[32 + lane],
                                                       (short)0, c, false, false);
            float s = 0.0f;
#pragma unroll
            for (int i = 0; i < 8; ++i) s += c[i];
#pragma unroll
            for (int off = 16; off > 0; off >>= 1) s += __shfl_xor(s, off, 32);
            if (lane == 0) schunk = s * (1.0f / 16.0f);
        }
        __syncthreads();

        if (flag) {
            const int myslot = carry + (int)swave[wave] + (incl - flag);
            if (myslot < MAX_VOX_) slot[b * NVOX_ + fl] = myslot;
        }
        carry += (int)(schunk + 0.5f);
        __syncthreads();
    }
}

// ---------------- voxel metadata: coords row [b,x,y,z] and npts -----------
__global__ void voxmeta_kernel(const int* __restrict__ slot, const int* __restrict__ cnt,
                               int* __restrict__ outc, int* __restrict__ outn) {
    const int i = blockIdx.x * blockDim.x + threadIdx.x;
    if (i >= B_ * NVOX_) return;
    const int s = slot[i];
    if (s < 0) return;
    const int b = i / NVOX_;
    const int v = i - b * NVOX_;
    const int iz = v % GZ_;
    const int r  = v / GZ_;
    const int iy = r % GY_;
    const int ix = r / GY_;
    const int o = (b * MAX_VOX_ + s) * 4;
    outc[o + 0] = b;
    outc[o + 1] = ix;
    outc[o + 2] = iy;
    outc[o + 3] = iz;
    const int c = cnt[i];
    outn[b * MAX_VOX_ + s] = c < MAX_PTS_ ? c : MAX_PTS_;
}

// ---------------- rank tournament: round r picks min unassigned point -----
__global__ void rank_min_kernel(const int* __restrict__ flat, const int* __restrict__ rank,
                                const int* __restrict__ slot, int* __restrict__ tmpm) {
    const long long idx = (long long)blockIdx.x * blockDim.x + threadIdx.x;
    if (idx >= (long long)B_ * N_) return;
    const int f = flat[idx];
    if (f < 0) return;
    if (rank[idx] >= 0) return;
    const int b = (int)(idx / N_);
    const int vi = b * NVOX_ + f;
    if (slot[vi] < 0) return;
    atomicMin(&tmpm[vi], (int)(idx - (long long)b * N_));
}

__global__ void rank_commit_kernel(const float* __restrict__ pts,
                                   const int* __restrict__ flat, int* __restrict__ rank,
                                   const int* __restrict__ slot, int* __restrict__ tmpm,
                                   float* __restrict__ outp, int r) {
    const long long idx = (long long)blockIdx.x * blockDim.x + threadIdx.x;
    if (idx >= (long long)B_ * N_) return;
    const int f = flat[idx];
    if (f < 0) return;
    if (rank[idx] >= 0) return;
    const int b = (int)(idx / N_);
    const int p = (int)(idx - (long long)b * N_);
    const int vi = b * NVOX_ + f;
    const int s = slot[vi];
    if (s < 0) return;
    if (tmpm[vi] != p) return;              // unique winner this round
    rank[idx] = r;
    tmpm[vi] = 0x7FFFFFFF;                  // reset for next round
    const float4 v = *(const float4*)(pts + idx * 4);
    float4* dst = (float4*)(outp + ((long long)(b * MAX_VOX_ + s) * MAX_PTS_ + r) * 4);
    *dst = v;
}

// ---------------- host-side launch --------------------------------------
extern "C" void kernel_launch(void* const* d_in, const int* in_sizes, int n_in,
                              void* d_out, int out_size, void* d_ws, size_t ws_size,
                              hipStream_t stream) {
    const float* pts = (const float*)d_in[0];
    float* out = (float*)d_out;

    const long long PILL = (long long)B_ * MAX_VOX_ * MAX_PTS_ * C_;
    const long long COOR = PILL + (long long)B_ * MAX_VOX_ * 4;
    int* outc = (int*)d_out + PILL;
    int* outn = (int*)d_out + COOR;

    int* ws     = (int*)d_ws;
    int* flat   = ws;                       // B_*N_
    int* rank   = flat + B_ * N_;           // B_*N_
    int* firstp = rank + B_ * N_;           // B_*NVOX_
    int* cnt    = firstp + B_ * NVOX_;      // B_*NVOX_
    int* slot   = cnt + B_ * NVOX_;         // B_*NVOX_
    int* tmpm   = slot + B_ * NVOX_;        // B_*NVOX_

    init_out_kernel<<<4096, 256, 0, stream>>>(out);
    init_ws_kernel<<<(B_ * NVOX_ + 255) / 256, 256, 0, stream>>>(firstp, cnt, slot, tmpm);

    const int nPtBlocks = (B_ * N_) / 256;  // 1875, exact
    flat_kernel<<<nPtBlocks, 256, 0, stream>>>(pts, flat, rank, firstp, cnt);

    slot_kernel<<<B_, 1024, 0, stream>>>(flat, firstp, slot);

    voxmeta_kernel<<<(B_ * NVOX_ + 255) / 256, 256, 0, stream>>>(slot, cnt, outc, outn);

    for (int r = 0; r < MAX_PTS_; ++r) {
        rank_min_kernel<<<nPtBlocks, 256, 0, stream>>>(flat, rank, slot, tmpm);
        rank_commit_kernel<<<nPtBlocks, 256, 0, stream>>>(pts, flat, rank, slot, tmpm, out, r);
    }
}
